// pcseq_classifier_36352603193512
// MI455X (gfx1250) — compile-verified, matrix operand order checked by reading, not run on hardware
//
#include <hip/hip_runtime.h>
#include <math.h>

// ---------------------------------------------------------------------------
// Problem constants (from reference)
// ---------------------------------------------------------------------------
#define B_   16
#define F_   60
#define N_   512
#define EMB_ 128
#define H_   4
#define DH_  32
#define NC_  27
#define L_   2
#define S_   61
#define SP_  64
#define FFD_ 256
#define CHUNK_ 128

typedef __attribute__((ext_vector_type(16))) _Float16 v16h;
typedef __attribute__((ext_vector_type(8)))  float    v8f;
typedef unsigned int v4u __attribute__((ext_vector_type(4)));
typedef int          v8i __attribute__((ext_vector_type(8)));
typedef int          v4i __attribute__((ext_vector_type(4)));

#define WMMA_F16(a, b, c) \
  __builtin_amdgcn_wmma_f32_16x16x32_f16(false, (a), false, (b), (short)0, (c), false, false)

// ---------------------------------------------------------------------------
// WMMA fragment helpers (CDNA5 ISA §7.12.2, wave32)
// ---------------------------------------------------------------------------
template <typename SrcT>
__device__ __forceinline__ v16h load_frag(const SrcT* base, int ld, int row0, int k0) {
  const int lane  = threadIdx.x & 31;
  const int m     = row0 + (lane & 15);
  const int khalf = (lane & 16) ? 8 : 0;
  const SrcT* p = base + (size_t)m * ld + k0 + khalf;
  v16h a;
#pragma unroll
  for (int i = 0; i < 8; ++i) a[i] = (_Float16)p[i];
#pragma unroll
  for (int i = 0; i < 8; ++i) a[8 + i] = (_Float16)p[16 + i];
  return a;
}

__device__ __forceinline__ void cframe(int& m0, int& n, int row0, int col0) {
  const int lane = threadIdx.x & 31;
  n  = col0 + (lane & 15);
  m0 = row0 + ((lane & 16) ? 8 : 0);
}

// ---------------------------------------------------------------------------
// Tensor Data Mover: 2D fp16 tile global -> LDS (CDNA5 ISA)
// D# group0: count=1, lds_addr[63:32], global_addr[120:64], type[127:126]=2
// D# group1: data_size=2B, tensor_dim0/1, tile_dim0/1 = whole tensor, stride0=dim0
// amdgpu-toolchain (clang-23) 6-arg builtin form.
// ---------------------------------------------------------------------------
__device__ __forceinline__ unsigned lds_off_of(const void* p) {
  return (unsigned)(unsigned long long)(const __attribute__((address_space(3))) char*)p;
}

__device__ __forceinline__ void tdm_load_2d(unsigned lds_off, const void* gptr,
                                            unsigned dim0 /*halves per row*/,
                                            unsigned dim1 /*rows*/) {
  unsigned long long ga = (unsigned long long)gptr;
  v4u g0;
  g0[0] = 1u;                                            // count=1 (valid), user mode
  g0[1] = lds_off;                                       // lds_addr (bytes)
  g0[2] = (unsigned)(ga & 0xFFFFFFFFu);                  // global_addr[31:0]
  g0[3] = (unsigned)((ga >> 32) & 0x01FFFFFFu) | 0x80000000u; // addr[56:32] | type=2
  v8i g1;
  g1[0] = 0x00010000;                                    // data_size=1 -> 2 bytes
  g1[1] = (int)((dim0 & 0xFFFFu) << 16);                 // tensor_dim0[15:0] @bit48
  g1[2] = (int)(((dim0 >> 16) & 0xFFFFu) | ((dim1 & 0xFFFFu) << 16)); // dim0 hi | dim1 lo
  g1[3] = (int)(((dim1 >> 16) & 0xFFFFu) | ((dim0 & 0xFFFFu) << 16)); // dim1 hi | tile_dim0
  g1[4] = (int)(dim1 & 0xFFFFu);                         // tile_dim1, tile_dim2=0
  g1[5] = (int)dim0;                                     // tensor_dim0_stride lo32
  g1[6] = 0;                                             // stride0 hi | stride1 lo
  g1[7] = 0;
  v4i z4 = {0, 0, 0, 0};
  v8i z8 = {0, 0, 0, 0, 0, 0, 0, 0};
  __builtin_amdgcn_tensor_load_to_lds(g0, g1, z4, z4, z8, 0);
}

__device__ __forceinline__ void layer_norm_row(float* row, const float* g, const float* bta) {
  float m = 0.0f;
  for (int i = 0; i < EMB_; ++i) m += row[i];
  m *= (1.0f / EMB_);
  float v = 0.0f;
  for (int i = 0; i < EMB_; ++i) { float d = row[i] - m; v += d * d; }
  v *= (1.0f / EMB_);
  float inv = rsqrtf(v + 1e-5f);
  for (int i = 0; i < EMB_; ++i) row[i] = (row[i] - m) * inv * g[i] + bta[i];
}

// ---------------------------------------------------------------------------
// Kernel 1: weight prep — FP32 -> FP16, (N,K)-major for WMMA B operands
// ---------------------------------------------------------------------------
__global__ void prep_kernel(const float* __restrict__ enc_w2, const float* __restrict__ enc_w3,
                            const float* __restrict__ in_w, const float* __restrict__ out_w,
                            const float* __restrict__ ff1_w, const float* __restrict__ ff2_w,
                            _Float16* __restrict__ w2t, _Float16* __restrict__ w3t,
                            _Float16* __restrict__ inwh, _Float16* __restrict__ outwh,
                            _Float16* __restrict__ ff1h, _Float16* __restrict__ ff2h) {
  const int i      = blockIdx.x * blockDim.x + threadIdx.x;
  const int stride = gridDim.x * blockDim.x;
  for (int j = i; j < 128 * 64; j += stride) {          // enc_w2^T
    int n = j >> 6, k = j & 63;
    w2t[j] = (_Float16)enc_w2[k * 128 + n];
  }
  for (int j = i; j < 128 * 128; j += stride) {         // enc_w3^T
    int n = j >> 7, k = j & 127;
    w3t[j] = (_Float16)enc_w3[k * 128 + n];
  }
  for (int j = i; j < L_ * 384 * 128; j += stride) inwh[j]  = (_Float16)in_w[j];
  for (int j = i; j < L_ * 128 * 128; j += stride) outwh[j] = (_Float16)out_w[j];
  for (int j = i; j < L_ * 256 * 128; j += stride) ff1h[j]  = (_Float16)ff1_w[j];
  for (int j = i; j < L_ * 128 * 256; j += stride) ff2h[j]  = (_Float16)ff2_w[j];
}

// ---------------------------------------------------------------------------
// Kernel 2: point encoder. TDM stages weights into LDS (overlapped with the
// VALU 3->64 layer), then paired-tile WMMA for 64->128 and 128->128.
// Dynamic LDS = 104 KB -> 3 workgroups per 320 KB WGP.
// ---------------------------------------------------------------------------
#define ENC_SMEM 106496

__global__ __launch_bounds__(256) void enc_kernel(
    const float* __restrict__ x,
    const float* __restrict__ w1, const float* __restrict__ b1,
    const _Float16* __restrict__ w2t, const float* __restrict__ b2,
    const _Float16* __restrict__ w3t, const float* __restrict__ b3,
    float* __restrict__ pmax) {
  extern __shared__ char esm[];
  _Float16* h1  = (_Float16*)esm;              // 128x64  f16  16384
  _Float16* h2  = (_Float16*)(esm + 16384);    // 128x128 f16  32768
  float*    mp  = (float*)(esm + 49152);       // 16x128  f32   8192
  _Float16* w2s = (_Float16*)(esm + 57344);    // 128x64  f16  16384
  _Float16* w3s = (_Float16*)(esm + 73728);    // 128x128 f16  32768

  const int chunk = blockIdx.x;
  const int f     = blockIdx.y;
  const int b     = blockIdx.z;
  const int tid   = threadIdx.x;
  const int wave  = tid >> 5;

  // async: DMA both weight matrices into LDS while layer 1 runs on VALU
  if (wave == 0) {
    tdm_load_2d(lds_off_of(w2s), w2t, 64, 128);
    tdm_load_2d(lds_off_of(w3s), w3t, 128, 128);
  }

  const float* xp = x + ((size_t)(b * F_ + f) * N_ + chunk * CHUNK_) * 3;

  // layer 1: relu(x @ W1 + b1), K=3 -> plain FMA
  for (int idx = tid; idx < CHUNK_ * 64; idx += 256) {
    int p = idx >> 6, o = idx & 63;
    float x0 = xp[p * 3 + 0], x1 = xp[p * 3 + 1], x2 = xp[p * 3 + 2];
    float v = x0 * w1[o] + x1 * w1[64 + o] + x2 * w1[128 + o] + b1[o];
    h1[p * 64 + o] = (_Float16)fmaxf(v, 0.0f);
  }
  if (wave == 0) __builtin_amdgcn_s_wait_tensorcnt(0);
  __syncthreads();

  // layer 2: relu(h1 @ W2 + b2)  M=128 N=128 K=64, N-tile pairs (indep. chains)
  for (int p = wave; p < 32; p += 8) {
    int tm = p >> 2, tn0 = (p & 3) * 2;
    v8f c0 = {}, c1 = {};
#pragma unroll
    for (int k = 0; k < 64; k += 32) {
      v16h a  = load_frag(h1, 64, tm * 16, k);
      v16h b0 = load_frag(w2s, 64, tn0 * 16, k);
      v16h b1v = load_frag(w2s, 64, tn0 * 16 + 16, k);
      c0 = WMMA_F16(a, b0, c0);
      c1 = WMMA_F16(a, b1v, c1);
    }
    int m0, n; cframe(m0, n, tm * 16, tn0 * 16);
    float bv0 = b2[n], bv1 = b2[n + 16];
#pragma unroll
    for (int r = 0; r < 8; ++r) {
      h2[(m0 + r) * 128 + n]      = (_Float16)fmaxf(c0[r] + bv0, 0.0f);
      h2[(m0 + r) * 128 + n + 16] = (_Float16)fmaxf(c1[r] + bv1, 0.0f);
    }
  }
  __syncthreads();

  // layer 3: h2 @ W3 + b3 with fused per-column max over points
  for (int p = wave; p < 32; p += 8) {
    int tm = p >> 2, tn0 = (p & 3) * 2;
    v8f c0 = {}, c1 = {};
#pragma unroll
    for (int k = 0; k < 128; k += 32) {
      v16h a  = load_frag(h2, 128, tm * 16, k);
      v16h b0 = load_frag(w3s, 128, tn0 * 16, k);
      v16h b1v = load_frag(w3s, 128, tn0 * 16 + 16, k);
      c0 = WMMA_F16(a, b0, c0);
      c1 = WMMA_F16(a, b1v, c1);
    }
    int m0, n; cframe(m0, n, tm * 16, tn0 * 16);
    float bv0 = b3[n], bv1 = b3[n + 16];
    float mx0 = -3.0e38f, mx1 = -3.0e38f;
#pragma unroll
    for (int r = 0; r < 8; ++r) {
      mx0 = fmaxf(mx0, c0[r] + bv0);
      mx1 = fmaxf(mx1, c1[r] + bv1);
    }
    int slot = tm * 2 + ((threadIdx.x & 16) ? 1 : 0);
    mp[slot * 128 + n]      = mx0;
    mp[slot * 128 + n + 16] = mx1;
  }
  __syncthreads();

  if (tid < 128) {
    float mx = mp[tid];
#pragma unroll
    for (int r = 1; r < 16; ++r) mx = fmaxf(mx, mp[r * 128 + tid]);
    pmax[((size_t)(b * F_ + f) * 4 + chunk) * 128 + tid] = mx;
  }
}

// ---------------------------------------------------------------------------
// Kernel 3: token assembly (cls+pos, masked maxpool + sinusoidal PE)
// ---------------------------------------------------------------------------
__global__ void ybuild_kernel(const float* __restrict__ pmax, const int* __restrict__ flen,
                              const float* __restrict__ cls_token,
                              const float* __restrict__ cls_pos, float* __restrict__ y) {
  const int s = blockIdx.x, b = blockIdx.y, c = threadIdx.x;
  float v;
  if (s == 0) {
    v = cls_token[c] + cls_pos[c];
  } else if (s <= F_) {
    const int f = s - 1;
    float mx = pmax[((size_t)(b * F_ + f) * 4 + 0) * 128 + c];
    for (int ch = 1; ch < 4; ++ch)
      mx = fmaxf(mx, pmax[((size_t)(b * F_ + f) * 4 + ch) * 128 + c]);
    float base = (f < flen[b]) ? mx : 0.0f;
    int j = c >> 1;
    float div = expf(-0.07195578829571529f * (float)(2 * j)); // ln(1e4)/128
    float ang = (float)f * div;
    v = base + ((c & 1) ? cosf(ang) : sinf(ang));
  } else {
    v = 0.0f;
  }
  y[((size_t)b * SP_ + s) * EMB_ + c] = v;
}

// ---------------------------------------------------------------------------
// Kernel 4: transformer, both layers in LDS; TDM prefetches out_w into the
// idle FF buffer during attention. Dynamic LDS = 152.25 KB (2 WGs/WGP).
// ---------------------------------------------------------------------------
#define XF_SMEM 155904

__device__ __forceinline__ void store_qkv(_Float16* qh, _Float16* kh, _Float16* vt,
                                          int m0, int n, const v8f& c, float bv,
                                          float qscale) {
  if (n < 128) {
#pragma unroll
    for (int r = 0; r < 8; ++r) qh[(m0 + r) * EMB_ + n] = (_Float16)((c[r] + bv) * qscale);
  } else if (n < 256) {
    int nn = n - 128;
#pragma unroll
    for (int r = 0; r < 8; ++r) kh[(m0 + r) * EMB_ + nn] = (_Float16)(c[r] + bv);
  } else {
    int d = n - 256;
#pragma unroll
    for (int r = 0; r < 8; ++r) vt[d * SP_ + (m0 + r)] = (_Float16)(c[r] + bv);
  }
}

__global__ __launch_bounds__(256) void xformer_kernel(
    float* __restrict__ y, const int* __restrict__ flen,
    const _Float16* __restrict__ inw,  const float* __restrict__ inb,
    const _Float16* __restrict__ outw, const float* __restrict__ outb,
    const float* __restrict__ ln1g, const float* __restrict__ ln1b,
    const _Float16* __restrict__ ff1w, const float* __restrict__ ff1b,
    const _Float16* __restrict__ ff2w, const float* __restrict__ ff2b,
    const float* __restrict__ ln2g, const float* __restrict__ ln2b) {
  extern __shared__ char smem[];
  float*    y_s  = (float*)smem;                    // 64x128 f32   32768
  _Float16* qh   = (_Float16*)(smem + 32768);       // 64x128 f16   16384
  _Float16* kh   = (_Float16*)(smem + 49152);       // 64x128 f16   16384
  _Float16* vt   = (_Float16*)(smem + 65536);       // 128x64 f16   16384 (v^T)
  float*    sc   = (float*)(smem + 81920);          // 64x64  f32   16384
  _Float16* ah   = (_Float16*)(smem + 98304);       // 64x64  f16    8192
  _Float16* oh   = (_Float16*)(smem + 106496);      // 64x128 f16   16384
  _Float16* fh   = (_Float16*)(smem + 122880);      // 64x256 f16   32768 (FF / staged out_w)
  float*    bias = (float*)(smem + 155648);         // 64     f32     256

  const int b    = blockIdx.x;
  const int tid  = threadIdx.x;
  const int wave = tid >> 5;
  const float qscale = 0.17677669529663687f;        // 1/sqrt(dh)

  __builtin_prefetch(inw, 0, 2);
  __builtin_prefetch(ff1w, 0, 2);

  for (int i = tid; i < SP_ * EMB_; i += 256) y_s[i] = y[(size_t)b * SP_ * EMB_ + i];
  if (tid < SP_) bias[tid] = (tid >= flen[b] + 1) ? -1.0e9f : 0.0f;
  __syncthreads();

  for (int l = 0; l < L_; ++l) {
    const _Float16* winl = inw + (size_t)l * 384 * EMB_;
    const float*    binl = inb + l * 384;

    // async: stage this layer's out_w into fh (idle until out-proj)
    if (wave == 0) tdm_load_2d(lds_off_of(fh), outw + (size_t)l * EMB_ * EMB_, EMB_, EMB_);

    // ---- QKV: (64x128)@(128x384), N-tile pairs ----
    for (int p = wave; p < 48; p += 8) {
      int tm = p / 12, tn0 = (p % 12) * 2;
      v8f c0 = {}, c1 = {};
#pragma unroll
      for (int k = 0; k < EMB_; k += 32) {
        v16h a  = load_frag(y_s, EMB_, tm * 16, k);
        v16h b0 = load_frag(winl, EMB_, tn0 * 16, k);
        v16h b1v = load_frag(winl, EMB_, tn0 * 16 + 16, k);
        c0 = WMMA_F16(a, b0, c0);
        c1 = WMMA_F16(a, b1v, c1);
      }
      int m0, n; cframe(m0, n, tm * 16, tn0 * 16);
      store_qkv(qh, kh, vt, m0, n, c0, binl[n], qscale);
      store_qkv(qh, kh, vt, m0, n + 16, c1, binl[n + 16], qscale);
    }
    __syncthreads();

    // ---- attention ----
    for (int h = 0; h < H_; ++h) {
      for (int p = wave; p < 8; p += 8) {             // scores, K=32, tile pairs
        int tm = p >> 1, tn0 = (p & 1) * 2;
        v8f c0 = {}, c1 = {};
        v16h a  = load_frag(qh, EMB_, tm * 16, h * DH_);
        v16h b0 = load_frag(kh, EMB_, tn0 * 16, h * DH_);
        v16h b1v = load_frag(kh, EMB_, tn0 * 16 + 16, h * DH_);
        c0 = WMMA_F16(a, b0, c0);
        c1 = WMMA_F16(a, b1v, c1);
        int m0, n; cframe(m0, n, tm * 16, tn0 * 16);
        float kb0 = bias[n], kb1 = bias[n + 16];
#pragma unroll
        for (int r = 0; r < 8; ++r) {
          sc[(m0 + r) * SP_ + n]      = c0[r] + kb0;
          sc[(m0 + r) * SP_ + n + 16] = c1[r] + kb1;
        }
      }
      __syncthreads();
      if (tid < SP_) {                                // row softmax -> f16
        float mx = -3.0e38f;
        for (int j = 0; j < SP_; ++j) mx = fmaxf(mx, sc[tid * SP_ + j]);
        float sum = 0.0f;
        for (int j = 0; j < SP_; ++j) {
          float e = expf(sc[tid * SP_ + j] - mx);
          sc[tid * SP_ + j] = e;
          sum += e;
        }
        float inv = 1.0f / sum;
        for (int j = 0; j < SP_; ++j) ah[tid * SP_ + j] = (_Float16)(sc[tid * SP_ + j] * inv);
      }
      __syncthreads();
      for (int t = wave; t < 8; t += 8) {             // o = attn @ v, K=64
        int tm = t >> 1, tn = t & 1;
        v8f c = {};
#pragma unroll
        for (int k = 0; k < SP_; k += 32) {
          v16h a  = load_frag(ah, SP_, tm * 16, k);
          v16h bb = load_frag(vt, SP_, h * DH_ + tn * 16, k);
          c = WMMA_F16(a, bb, c);
        }
        int m0, n; cframe(m0, n, tm * 16, tn * 16);
#pragma unroll
        for (int r = 0; r < 8; ++r) oh[(m0 + r) * EMB_ + h * DH_ + n] = (_Float16)c[r];
      }
      __syncthreads();
    }

    // ---- out proj (B from TDM-staged LDS) + residual ----
    if (wave == 0) __builtin_amdgcn_s_wait_tensorcnt(0);
    __syncthreads();
    for (int p = wave; p < 16; p += 8) {
      int tm = p >> 2, tn0 = (p & 3) * 2;
      v8f c0 = {}, c1 = {};
#pragma unroll
      for (int k = 0; k < EMB_; k += 32) {
        v16h a  = load_frag(oh, EMB_, tm * 16, k);
        v16h b0 = load_frag(fh, EMB_, tn0 * 16, k);
        v16h b1v = load_frag(fh, EMB_, tn0 * 16 + 16, k);
        c0 = WMMA_F16(a, b0, c0);
        c1 = WMMA_F16(a, b1v, c1);
      }
      int m0, n; cframe(m0, n, tm * 16, tn0 * 16);
      float bv0 = outb[l * EMB_ + n], bv1 = outb[l * EMB_ + n + 16];
#pragma unroll
      for (int r = 0; r < 8; ++r) {
        y_s[(m0 + r) * EMB_ + n]      += c0[r] + bv0;
        y_s[(m0 + r) * EMB_ + n + 16] += c1[r] + bv1;
      }
    }
    __syncthreads();
    if (tid < SP_) layer_norm_row(y_s + tid * EMB_, ln1g + l * EMB_, ln1b + l * EMB_);
    __syncthreads();

    // ---- FF1 relu -> fh ----
    const _Float16* f1 = ff1w + (size_t)l * FFD_ * EMB_;
    for (int p = wave; p < 32; p += 8) {
      int tm = p >> 3, tn0 = (p & 7) * 2;
      v8f c0 = {}, c1 = {};
#pragma unroll
      for (int k = 0; k < EMB_; k += 32) {
        v16h a  = load_frag(y_s, EMB_, tm * 16, k);
        v16h b0 = load_frag(f1, EMB_, tn0 * 16, k);
        v16h b1v = load_frag(f1, EMB_, tn0 * 16 + 16, k);
        c0 = WMMA_F16(a, b0, c0);
        c1 = WMMA_F16(a, b1v, c1);
      }
      int m0, n; cframe(m0, n, tm * 16, tn0 * 16);
      float bv0 = ff1b[l * FFD_ + n], bv1 = ff1b[l * FFD_ + n + 16];
#pragma unroll
      for (int r = 0; r < 8; ++r) {
        fh[(m0 + r) * FFD_ + n]      = (_Float16)fmaxf(c0[r] + bv0, 0.0f);
        fh[(m0 + r) * FFD_ + n + 16] = (_Float16)fmaxf(c1[r] + bv1, 0.0f);
      }
    }
    __syncthreads();

    // ---- FF2 + residual ----
    const _Float16* f2 = ff2w + (size_t)l * EMB_ * FFD_;
    for (int p = wave; p < 16; p += 8) {
      int tm = p >> 2, tn0 = (p & 3) * 2;
      v8f c0 = {}, c1 = {};
#pragma unroll
      for (int k = 0; k < FFD_; k += 32) {
        v16h a  = load_frag(fh, FFD_, tm * 16, k);
        v16h b0 = load_frag(f2, FFD_, tn0 * 16, k);
        v16h b1v = load_frag(f2, FFD_, tn0 * 16 + 16, k);
        c0 = WMMA_F16(a, b0, c0);
        c1 = WMMA_F16(a, b1v, c1);
      }
      int m0, n; cframe(m0, n, tm * 16, tn0 * 16);
      float bv0 = ff2b[l * EMB_ + n], bv1 = ff2b[l * EMB_ + n + 16];
#pragma unroll
      for (int r = 0; r < 8; ++r) {
        y_s[(m0 + r) * EMB_ + n]      += c0[r] + bv0;
        y_s[(m0 + r) * EMB_ + n + 16] += c1[r] + bv1;
      }
    }
    __syncthreads();
    if (tid < SP_) layer_norm_row(y_s + tid * EMB_, ln2g + l * EMB_, ln2b + l * EMB_);
    __syncthreads();
  }

  for (int i = tid; i < SP_ * EMB_; i += 256) y[(size_t)b * SP_ * EMB_ + i] = y_s[i];
}

// ---------------------------------------------------------------------------
// Kernel 5: classifier head (tiny) — fc/BN/leakyReLU chain in one block.
// ---------------------------------------------------------------------------
__device__ __forceinline__ void bn_col(const float* base, int ld, int c, int rows,
                                       float g, float bb, float* dst) {
  float m = 0.0f;
  for (int r = 0; r < rows; ++r) m += base[r * ld + c];
  m /= (float)rows;
  float v = 0.0f;
  for (int r = 0; r < rows; ++r) { float d = base[r * ld + c] - m; v += d * d; }
  v /= (float)rows;
  float inv = rsqrtf(v + 1e-5f);
  for (int r = 0; r < rows; ++r) {
    float t = (base[r * ld + c] - m) * inv * g + bb;
    dst[r * ld + c] = (t > 0.0f) ? t : 0.2f * t;
  }
}

__global__ __launch_bounds__(256) void head_kernel(
    const float* __restrict__ y, const int* __restrict__ flen,
    const float* fc1w, const float* fc1b, const float* bn1g, const float* bn1b,
    const float* fc15w, const float* fc15b, const float* bn15g, const float* bn15b,
    const float* fc2w, const float* fc2b, const float* bn2g, const float* bn2b,
    const float* fcw, const float* fcb, float* __restrict__ out) {
  __shared__ float zc[B_][256];
  __shared__ float t1[B_][128], a1[B_][128];
  __shared__ float t2[B_][64],  a2[B_][64];
  __shared__ float t3[B_][32],  a3[B_][32];
  const int tid = threadIdx.x;

  for (int i = tid; i < B_ * 256; i += 256) {
    int bb = i >> 8, c = i & 255;
    zc[bb][c] = (c < 128) ? y[(size_t)bb * SP_ * EMB_ + c]
                          : y[(size_t)bb * SP_ * EMB_ + (size_t)flen[bb] * EMB_ + (c - 128)];
  }
  __syncthreads();
  for (int i = tid; i < B_ * 128; i += 256) {
    int bb = i >> 7, o = i & 127;
    float acc = fc1b[o];
    for (int k = 0; k < 256; ++k) acc += zc[bb][k] * fc1w[k * 128 + o];
    t1[bb][o] = acc;
  }
  __syncthreads();
  if (tid < 128) bn_col(&t1[0][0], 128, tid, B_, bn1g[tid], bn1b[tid], &a1[0][0]);
  __syncthreads();
  for (int i = tid; i < B_ * 64; i += 256) {
    int bb = i >> 6, o = i & 63;
    float acc = fc15b[o];
    for (int k = 0; k < 128; ++k) acc += a1[bb][k] * fc15w[k * 64 + o];
    t2[bb][o] = acc;
  }
  __syncthreads();
  if (tid < 64) bn_col(&t2[0][0], 64, tid, B_, bn15g[tid], bn15b[tid], &a2[0][0]);
  __syncthreads();
  for (int i = tid; i < B_ * 32; i += 256) {
    int bb = i >> 5, o = i & 31;
    float acc = fc2b[o];
    for (int k = 0; k < 64; ++k) acc += a2[bb][k] * fc2w[k * 32 + o];
    t3[bb][o] = acc;
  }
  __syncthreads();
  if (tid < 32) bn_col(&t3[0][0], 32, tid, B_, bn2g[tid], bn2b[tid], &a3[0][0]);
  __syncthreads();
  for (int i = tid; i < B_ * NC_; i += 256) {
    int bb = i / NC_, o = i % NC_;
    float acc = fcb[o];
    for (int k = 0; k < 32; ++k) acc += a3[bb][k] * fcw[k * NC_ + o];
    out[i] = acc;
  }
}

// ---------------------------------------------------------------------------
// Host launcher
// ---------------------------------------------------------------------------
extern "C" void kernel_launch(void* const* d_in, const int* in_sizes, int n_in,
                              void* d_out, int out_size, void* d_ws, size_t ws_size,
                              hipStream_t stream) {
  (void)in_sizes; (void)n_in; (void)out_size; (void)ws_size;
  const float* x         = (const float*)d_in[0];
  const int*   flen      = (const int*)d_in[1];
  const float* enc_w1    = (const float*)d_in[2];
  const float* enc_b1    = (const float*)d_in[3];
  const float* enc_w2    = (const float*)d_in[4];
  const float* enc_b2    = (const float*)d_in[5];
  const float* enc_w3    = (const float*)d_in[6];
  const float* enc_b3    = (const float*)d_in[7];
  const float* cls_token = (const float*)d_in[8];
  const float* cls_pos   = (const float*)d_in[9];
  const float* in_w      = (const float*)d_in[10];
  const float* in_b      = (const float*)d_in[11];
  const float* out_w     = (const float*)d_in[12];
  const float* out_b     = (const float*)d_in[13];
  const float* ln1_g     = (const float*)d_in[14];
  const float* ln1_b     = (const float*)d_in[15];
  const float* ff1_w     = (const float*)d_in[16];
  const float* ff1_b     = (const float*)d_in[17];
  const float* ff2_w     = (const float*)d_in[18];
  const float* ff2_b     = (const float*)d_in[19];
  const float* ln2_g     = (const float*)d_in[20];
  const float* ln2_b     = (const float*)d_in[21];
  const float* fc1_w     = (const float*)d_in[22];
  const float* fc1_b     = (const float*)d_in[23];
  const float* bn1_g     = (const float*)d_in[24];
  const float* bn1_b     = (const float*)d_in[25];
  const float* fc15_w    = (const float*)d_in[26];
  const float* fc15_b    = (const float*)d_in[27];
  const float* bn15_g    = (const float*)d_in[28];
  const float* bn15_b    = (const float*)d_in[29];
  const float* fc2_w     = (const float*)d_in[30];
  const float* fc2_b     = (const float*)d_in[31];
  const float* bn2_g     = (const float*)d_in[32];
  const float* bn2_b     = (const float*)d_in[33];
  const float* fc_w      = (const float*)d_in[34];
  const float* fc_b      = (const float*)d_in[35];
  float* out = (float*)d_out;

  char* ws = (char*)d_ws;
  _Float16* w2t   = (_Float16*)(ws + 0);        //  16384
  _Float16* w3t   = (_Float16*)(ws + 16384);    //  32768
  _Float16* inwh  = (_Float16*)(ws + 49152);    // 196608
  _Float16* outwh = (_Float16*)(ws + 245760);   //  65536
  _Float16* ff1h  = (_Float16*)(ws + 311296);   // 131072
  _Float16* ff2h  = (_Float16*)(ws + 442368);   // 131072
  float*    pmax  = (float*)(ws + 573440);      // 1966080
  float*    ybuf  = (float*)(ws + 2539520);     // 524288  (total ~3.06 MB)

  prep_kernel<<<dim3(256), dim3(256), 0, stream>>>(
      enc_w2, enc_w3, in_w, out_w, ff1_w, ff2_w, w2t, w3t, inwh, outwh, ff1h, ff2h);

  enc_kernel<<<dim3(4, F_, B_), dim3(256), ENC_SMEM, stream>>>(
      x, enc_w1, enc_b1, w2t, enc_b2, w3t, enc_b3, pmax);

  ybuild_kernel<<<dim3(SP_, B_), dim3(128), 0, stream>>>(
      pmax, flen, cls_token, cls_pos, ybuf);

  xformer_kernel<<<dim3(B_), dim3(256), XF_SMEM, stream>>>(
      ybuf, flen, inwh, in_b, outwh, out_b, ln1_g, ln1_b,
      ff1h, ff1_b, ff2h, ff2_b, ln2_g, ln2_b);

  head_kernel<<<dim3(1), dim3(256), 0, stream>>>(
      ybuf, flen, fc1_w, fc1_b, bn1_g, bn1_b, fc15_w, fc15_b, bn15_g, bn15_b,
      fc2_w, fc2_b, bn2_g, bn2_b, fc_w, fc_b, out);
}
